// SimplifiedImageOnlyDecomposer_1400159338927
// MI455X (gfx1250) — matrix operand order repaired and checked by reading, not compiled, not run on hardware
//
#include <hip/hip_runtime.h>

typedef __attribute__((ext_vector_type(16))) __bf16 v16bf;
typedef __attribute__((ext_vector_type(8)))  float  v8f;

union ABf {
    v16bf v;
    unsigned short u[16];
    unsigned int d[8];
};

__device__ __forceinline__ unsigned short f2bf(float f) {
    unsigned int u = __builtin_bit_cast(unsigned int, f);
    u += 0x7FFFu + ((u >> 16) & 1u);     // round-to-nearest-even
    return (unsigned short)(u >> 16);
}

// ---------------------------------------------------------------------------
// f32 -> bf16 convert
// ---------------------------------------------------------------------------
__global__ __launch_bounds__(256)
void cvt_bf16_kernel(const float* __restrict__ src, unsigned short* __restrict__ dst, long n) {
    long i = (long)blockIdx.x * 256 + threadIdx.x;
    if (i < n) dst[i] = f2bf(src[i]);
}

// ---------------------------------------------------------------------------
// Per-(b,h) transpose of v: qkv f32 [n=576, dh=64] (stride 1536, offset 1024+h*64)
//  -> vT bf16 [dh=64, n=576]. LDS 64x64 tile transpose; grid (ntile=9, 1, 64).
// Makes the PV GEMM's B operand K-contiguous so it uses the b128 WMMA path.
// ---------------------------------------------------------------------------
__global__ __launch_bounds__(256)
void vtrans_kernel(const float* __restrict__ qkv, unsigned short* __restrict__ vT)
{
    __shared__ float tile[64][65];
    const int z  = blockIdx.z;           // b*8 + h
    const int bz = z >> 3, hz = z & 7;
    const int n0 = blockIdx.x * 64;

    const float* src = qkv + (long)bz * 576 * 1536 + 1024 + hz * 64;
    // read v[n0+nl][dh], coalesced along dh
    for (int idx = threadIdx.x; idx < 64 * 64; idx += 256) {
        const int nl = idx >> 6, dh = idx & 63;
        tile[nl][dh] = src[(long)(n0 + nl) * 1536 + dh];
    }
    __syncthreads();
    // write vT[dh][n0+nl], coalesced along nl
    unsigned short* dst = vT + (long)z * 64 * 576;
    for (int idx = threadIdx.x; idx < 64 * 64; idx += 256) {
        const int dh = idx >> 6, nl = idx & 63;
        dst[(long)dh * 576 + n0 + nl] = f2bf(tile[nl][dh]);
    }
}

// ---------------------------------------------------------------------------
// bf16 WMMA GEMM:  C[M,N] = scale * (A[M,K] @ W[N,K]^T) + bias + resid
// Both operands K-contiguous (row-major [rows, K]); all loads coalesce to b128.
// Batched over blockIdx.z = b*zdiv + h with independent (b,h) strides.
// Each wave computes one 16x16 tile via v_wmma_f32_16x16x32_bf16.
// ---------------------------------------------------------------------------
__global__ __launch_bounds__(256)
void gemm_bf16_wmma(const unsigned short* __restrict__ A, int lda,
                    const unsigned short* __restrict__ Bm, int ldb,
                    const float* __restrict__ bias,
                    const float* __restrict__ resid, int ldr,
                    float* __restrict__ C, int ldc,
                    int Mrows, int Ncols, int K, float scale,
                    int zdiv, long sAb, long sAh, long sBb, long sBh,
                    long sCb, long sCh)
{
    const int z = blockIdx.z;
    const int bz = z / zdiv, hz = z % zdiv;
    A  += (long)bz * sAb + (long)hz * sAh;
    Bm += (long)bz * sBb + (long)hz * sBh;
    C  += (long)bz * sCb + (long)hz * sCh;

    const int lane = threadIdx.x & 31;
    const int wave = threadIdx.x >> 5;
    const int ntiles = (Ncols + 15) >> 4;
    const int mtiles = (Mrows + 15) >> 4;
    const int tile = blockIdx.x * 8 + wave;
    if (tile >= ntiles * mtiles) return;          // wave-uniform
    const int tm = tile / ntiles;
    const int tn = tile % ntiles;

    const int row16 = lane & 15;
    const int half  = lane >> 4;
    const int arow  = tm * 16 + row16;
    const int bcol  = tn * 16 + row16;

    v8f acc = {};
    const unsigned short* Arow = A + (long)arow * lda;
    const unsigned short* Brow = Bm + (long)bcol * ldb;

    for (int k0 = 0; k0 < K; k0 += 32) {
        __builtin_prefetch((const void*)(Arow + k0 + 128), 0, 1);
        ABf a, b;
        // A fragment: 16-bit A 16x32 layout (M = lane&15, K groups select by lane>>4)
        const int ka = k0 + half * 8;
#pragma unroll
        for (int j = 0; j < 4; ++j) {
            a.d[j]     = *(const unsigned int*)(Arow + ka + 2 * j);
            a.d[4 + j] = *(const unsigned int*)(Arow + ka + 16 + 2 * j);
        }
        // B fragment: 16-bit B 32x16 layout (N = lane&15, K groups select by lane>>4, same as A)
        const int kb = k0 + half * 8;
#pragma unroll
        for (int j = 0; j < 4; ++j) {
            b.d[j]     = *(const unsigned int*)(Brow + kb + 2 * j);
            b.d[4 + j] = *(const unsigned int*)(Brow + kb + 16 + 2 * j);
        }

        acc = __builtin_amdgcn_wmma_f32_16x16x32_bf16(
            false, a.v, false, b.v, (short)0, acc, false, false);
    }

    // C/D layout: VGPR r -> M = tm*16 + half*8 + r, N = tn*16 + (lane&15)
#pragma unroll
    for (int r = 0; r < 8; ++r) {
        const int row = tm * 16 + half * 8 + r;
        const int col = tn * 16 + row16;
        if (row < Mrows && col < Ncols) {
            float v = acc[r] * scale;
            if (bias)  v += bias[col];
            if (resid) v += resid[(long)row * ldr + col];
            C[(long)row * ldc + col] = v;
        }
    }
}

// ---------------------------------------------------------------------------
// Row softmax (scores pre-scaled by `scale`), output bf16 probs.
// One block per row of `ncols`.
// ---------------------------------------------------------------------------
__global__ __launch_bounds__(256)
void softmax_kernel(const float* __restrict__ S, unsigned short* __restrict__ P,
                    int ncols, float scale)
{
    const long row = blockIdx.x;
    const float* s = S + row * (long)ncols;
    unsigned short* p = P + row * (long)ncols;
    __shared__ float red[256];

    float mx = -3.4e38f;
    for (int i = threadIdx.x; i < ncols; i += 256) mx = fmaxf(mx, s[i] * scale);
    red[threadIdx.x] = mx; __syncthreads();
    for (int o = 128; o > 0; o >>= 1) {
        if (threadIdx.x < o) red[threadIdx.x] = fmaxf(red[threadIdx.x], red[threadIdx.x + o]);
        __syncthreads();
    }
    mx = red[0]; __syncthreads();

    float sum = 0.f;
    for (int i = threadIdx.x; i < ncols; i += 256) sum += __expf(s[i] * scale - mx);
    red[threadIdx.x] = sum; __syncthreads();
    for (int o = 128; o > 0; o >>= 1) {
        if (threadIdx.x < o) red[threadIdx.x] += red[threadIdx.x + o];
        __syncthreads();
    }
    const float inv = 1.0f / red[0];
    for (int i = threadIdx.x; i < ncols; i += 256)
        p[i] = f2bf(__expf(s[i] * scale - mx) * inv);
}

// ---------------------------------------------------------------------------
// LayerNorm (D=512, eps=1e-5) + exact GELU, output bf16. One block per row.
// ---------------------------------------------------------------------------
__global__ __launch_bounds__(256)
void ln_gelu_kernel(const float* __restrict__ Hin,
                    const float* __restrict__ g, const float* __restrict__ bta,
                    unsigned short* __restrict__ Hout)
{
    const long row = blockIdx.x;
    const float* h = Hin + row * 512;
    __shared__ float red[256];

    const int t = threadIdx.x;
    const float x0 = h[t], x1 = h[t + 256];
    red[t] = x0 + x1; __syncthreads();
    for (int o = 128; o > 0; o >>= 1) { if (t < o) red[t] += red[t + o]; __syncthreads(); }
    const float mu = red[0] * (1.0f / 512.0f); __syncthreads();

    const float d0 = x0 - mu, d1 = x1 - mu;
    red[t] = d0 * d0 + d1 * d1; __syncthreads();
    for (int o = 128; o > 0; o >>= 1) { if (t < o) red[t] += red[t + o]; __syncthreads(); }
    const float inv = rsqrtf(red[0] * (1.0f / 512.0f) + 1e-5f);

    float y0 = d0 * inv * g[t]       + bta[t];
    float y1 = d1 * inv * g[t + 256] + bta[t + 256];
    y0 = 0.5f * y0 * (1.0f + erff(y0 * 0.70710678118654752f));
    y1 = 0.5f * y1 * (1.0f + erff(y1 * 0.70710678118654752f));
    Hout[row * 512 + t]       = f2bf(y0);
    Hout[row * 512 + t + 256] = f2bf(y1);
}

// ---------------------------------------------------------------------------
// Outer-product + per-(b,n,m) top-K(=51) sparsify.
// Ranking key |F[d]*T[m,d]*comp[m]| shares the per-row factor |comp[m]|, so the
// threshold is found on a[d] = |F[d]*T[m,d]| with a 31-step binary search over
// the (monotone) float bit pattern, one wave per m. Output writes coalesced.
// ---------------------------------------------------------------------------
__global__ __launch_bounds__(256)
void sparsify_kernel(const float* __restrict__ F, const float* __restrict__ T,
                     const float* __restrict__ comp, float* __restrict__ out)
{
    const long bn = blockIdx.x;                  // [0, B*N)
    __shared__ float sF[512];
    __shared__ float sT[16 * 512];
    __shared__ float sThr[16];
    __shared__ float sC[16];

    const float* Fr = F + bn * 512;
    for (int i = threadIdx.x; i < 512; i += 256)       sF[i] = Fr[i];
    for (int i = threadIdx.x; i < 16 * 512; i += 256)  sT[i] = T[i];
    if (threadIdx.x < 16) sC[threadIdx.x] = comp[bn * 16 + threadIdx.x];
    __syncthreads();

    const int wave = threadIdx.x >> 5, lane = threadIdx.x & 31;
    for (int m = wave; m < 16; m += 8) {
        float a[16];
#pragma unroll
        for (int j = 0; j < 16; ++j) {
            const int d = lane + j * 32;
            a[j] = fabsf(sF[d] * sT[m * 512 + d]);
        }
        // largest threshold t with count(a >= t) >= K  (K = 51)
        unsigned int t = 0;
        for (int bit = 30; bit >= 0; --bit) {
            const unsigned int cand = t | (1u << bit);
            const float cf = __builtin_bit_cast(float, cand);
            int cnt = 0;
#pragma unroll
            for (int j = 0; j < 16; ++j) cnt += (a[j] >= cf) ? 1 : 0;
            for (int o = 16; o > 0; o >>= 1) cnt += __shfl_xor(cnt, o, 32);
            if (cnt >= 51) t = cand;
        }
        if (lane == 0) sThr[m] = __builtin_bit_cast(float, t);
    }
    __syncthreads();

    float* o = out + bn * 8192;                  // D*M contiguous slab
    for (int i = threadIdx.x; i < 8192; i += 256) {
        const int d = i >> 4, m = i & 15;
        const float prod = sF[d] * sT[m * 512 + d];
        o[i] = (fabsf(prod) >= sThr[m]) ? prod * sC[m] : 0.0f;
    }
}

// ---------------------------------------------------------------------------
// Host launcher
// ---------------------------------------------------------------------------
extern "C" void kernel_launch(void* const* d_in, const int* in_sizes, int n_in,
                              void* d_out, int out_size, void* d_ws, size_t ws_size,
                              hipStream_t stream)
{
    const float* F    = (const float*)d_in[0];   // [8,576,512]
    const float* Win  = (const float*)d_in[1];   // [1536,512]
    const float* bin  = (const float*)d_in[2];   // [1536]
    const float* Wout = (const float*)d_in[3];   // [512,512]
    const float* bout = (const float*)d_in[4];   // [512]
    const float* W1   = (const float*)d_in[5];   // [512,512]
    const float* b1   = (const float*)d_in[6];   // [512]
    const float* lng  = (const float*)d_in[7];   // [512]
    const float* lnb  = (const float*)d_in[8];   // [512]
    const float* W2   = (const float*)d_in[9];   // [16,512]
    const float* b2   = (const float*)d_in[10];  // [16]
    const float* Tm   = (const float*)d_in[11];  // [16,512]
    float* out = (float*)d_out;

    const long BN = 4608;          // B*N
    const long S  = 576L * 576;    // scores per (b,h)

    char* ws = (char*)d_ws;
    size_t off = 0;
    auto alloc = [&](size_t bytes) -> char* {
        char* p = ws + off;
        off = (off + bytes + 255) & ~(size_t)255;
        return p;
    };
    unsigned short* F_bf    = (unsigned short*)alloc(BN * 512 * 2);
    unsigned short* Win_bf  = (unsigned short*)alloc(1536L * 512 * 2);
    unsigned short* Wout_bf = (unsigned short*)alloc(512L * 512 * 2);
    unsigned short* W1_bf   = (unsigned short*)alloc(512L * 512 * 2);
    unsigned short* W2_bf   = (unsigned short*)alloc(16L * 512 * 2);
    float*          qkv     = (float*)alloc(BN * 1536 * 4);
    unsigned short* qkv_bf  = (unsigned short*)alloc(BN * 1536 * 2);
    unsigned short* vT_bf   = (unsigned short*)alloc(64L * 64 * 576 * 2);
    float*          scores  = (float*)alloc(64L * S * 4);
    unsigned short* probs   = (unsigned short*)alloc(64L * S * 2);
    float*          o_f32   = (float*)alloc(BN * 512 * 4);
    unsigned short* o_bf    = (unsigned short*)alloc(BN * 512 * 2);
    float*          Fenh    = (float*)alloc(BN * 512 * 4);
    unsigned short* Fenh_bf = (unsigned short*)alloc(BN * 512 * 2);
    float*          h_f32   = (float*)alloc(BN * 512 * 4);
    unsigned short* h2_bf   = (unsigned short*)alloc(BN * 512 * 2);
    float*          compb   = (float*)alloc(BN * 16 * 4);

    auto cdiv = [](long a, long b) { return (int)((a + b - 1) / b); };

    // bf16 copies of F and weights
    cvt_bf16_kernel<<<cdiv(BN * 512, 256), 256, 0, stream>>>(F, F_bf, BN * 512);
    cvt_bf16_kernel<<<cdiv(1536L * 512, 256), 256, 0, stream>>>(Win, Win_bf, 1536L * 512);
    cvt_bf16_kernel<<<cdiv(512L * 512, 256), 256, 0, stream>>>(Wout, Wout_bf, 512L * 512);
    cvt_bf16_kernel<<<cdiv(512L * 512, 256), 256, 0, stream>>>(W1, W1_bf, 512L * 512);
    cvt_bf16_kernel<<<cdiv(16L * 512, 256), 256, 0, stream>>>(W2, W2_bf, 16L * 512);

    // qkv = F @ Win^T + bin                       [4608,1536]
    gemm_bf16_wmma<<<dim3(cdiv((4608 / 16) * (1536 / 16), 8), 1, 1), 256, 0, stream>>>(
        F_bf, 512, Win_bf, 512, bin, nullptr, 0, qkv, 1536,
        4608, 1536, 512, 1.0f, 1, 0, 0, 0, 0, 0, 0);
    cvt_bf16_kernel<<<cdiv(BN * 1536, 256), 256, 0, stream>>>(qkv, qkv_bf, BN * 1536);
    // v transposed per (b,h) -> bf16 [z][dh=64][n=576]
    vtrans_kernel<<<dim3(9, 1, 64), 256, 0, stream>>>(qkv, vT_bf);

    // scores(b,h) = q_h @ k_h^T                   64 x [576,576], K=64
    gemm_bf16_wmma<<<dim3(cdiv(36 * 36, 8), 1, 64), 256, 0, stream>>>(
        qkv_bf, 1536, qkv_bf + 512, 1536, nullptr, nullptr, 0, scores, 576,
        576, 576, 64, 1.0f,
        8, 576L * 1536, 64, 576L * 1536, 64, 8L * S, S);

    // softmax with 1/sqrt(64) scaling -> bf16 probs
    softmax_kernel<<<64 * 576, 256, 0, stream>>>(scores, probs, 576, 0.125f);

    // o(b,h) = probs @ v_h = probs @ (vT_h)^T     64 x [576,64], K=576
    gemm_bf16_wmma<<<dim3(cdiv(36 * 4, 8), 1, 64), 256, 0, stream>>>(
        probs, 576, vT_bf, 576, nullptr, nullptr, 0, o_f32, 512,
        576, 64, 576, 1.0f,
        8, 8L * S, S, 8L * 64 * 576, 64L * 576, 576L * 512, 64);
    cvt_bf16_kernel<<<cdiv(BN * 512, 256), 256, 0, stream>>>(o_f32, o_bf, BN * 512);

    // F_enh = F + (o @ Wout^T + bout)             [4608,512]
    gemm_bf16_wmma<<<dim3(cdiv((4608 / 16) * (512 / 16), 8), 1, 1), 256, 0, stream>>>(
        o_bf, 512, Wout_bf, 512, bout, F, 512, Fenh, 512,
        4608, 512, 512, 1.0f, 1, 0, 0, 0, 0, 0, 0);
    cvt_bf16_kernel<<<cdiv(BN * 512, 256), 256, 0, stream>>>(Fenh, Fenh_bf, BN * 512);

    // h = F_enh @ W1^T + b1                       [4608,512]
    gemm_bf16_wmma<<<dim3(cdiv((4608 / 16) * (512 / 16), 8), 1, 1), 256, 0, stream>>>(
        Fenh_bf, 512, W1_bf, 512, b1, nullptr, 0, h_f32, 512,
        4608, 512, 512, 1.0f, 1, 0, 0, 0, 0, 0, 0);

    // LayerNorm + exact GELU -> bf16
    ln_gelu_kernel<<<4608, 256, 0, stream>>>(h_f32, lng, lnb, h2_bf);

    // components = h2 @ W2^T + b2                 [4608,16]
    gemm_bf16_wmma<<<dim3(cdiv((4608 / 16) * (16 / 16), 8), 1, 1), 256, 0, stream>>>(
        h2_bf, 512, W2_bf, 512, b2, nullptr, 0, compb, 16,
        4608, 16, 512, 1.0f, 1, 0, 0, 0, 0, 0, 0);

    // outer product + top-K sparsify -> out [8,576,512,16]
    sparsify_kernel<<<4608, 256, 0, stream>>>(F, Tm, compb, out);
}
